// TripletTransformer_23021024706982
// MI455X (gfx1250) — matrix operand order repaired and checked by reading, not compiled, not run on hardware
//
#include <hip/hip_runtime.h>

#define N_NODES 20000
#define E_EDGES 320000
#define DMODEL  256
#define NHEADS  8
#define DHEAD   32

typedef __attribute__((ext_vector_type(16))) __bf16 v16bf;
typedef __attribute__((ext_vector_type(8)))  float  v8f;

// ---------- helpers ----------
__device__ __forceinline__ unsigned short f2bf(float f) {
    unsigned int u = __float_as_uint(f);
    unsigned int r = (u + 0x7FFFu + ((u >> 16) & 1u)) >> 16;
    return (unsigned short)r;
}

__device__ __forceinline__ float wave_sum32(float v) {
#pragma unroll
    for (int o = 16; o > 0; o >>= 1) v += __shfl_xor(v, o, 32);
    return v;
}

__device__ __forceinline__ float atomicMaxF(float* addr, float val) {
    int* ia = (int*)addr;
    int old = __float_as_int(*addr);
    while (__int_as_float(old) < val) {
        int assumed = old;
        old = atomicCAS(ia, assumed, __float_as_int(val));
        if (old == assumed) break;
    }
    return __int_as_float(old);
}

// ---------- elementwise kernels ----------
__global__ void __launch_bounds__(256) k_fill(float* __restrict__ p, float v, long long n) {
    long long i = (long long)blockIdx.x * blockDim.x + threadIdx.x;
    if (i < n) p[i] = v;
}

__global__ void __launch_bounds__(256) k_f32_to_bf16(const float* __restrict__ in,
                                                     unsigned short* __restrict__ out,
                                                     long long n) {
    long long i = (long long)blockIdx.x * blockDim.x + threadIdx.x;
    if (i < n) out[i] = f2bf(in[i]);
}

__global__ void __launch_bounds__(256) k_add(const float* __restrict__ a,
                                             const float* __restrict__ b,
                                             float* __restrict__ out, long long n) {
    long long i = (long long)blockIdx.x * blockDim.x + threadIdx.x;
    if (i < n) out[i] = a[i] + b[i];
}

__global__ void __launch_bounds__(256) k_add_inplace(float* __restrict__ out,
                                                     const float* __restrict__ a,
                                                     long long n) {
    long long i = (long long)blockIdx.x * blockDim.x + threadIdx.x;
    if (i < n) out[i] = out[i] + a[i];
}

__global__ void __launch_bounds__(256) k_gelu_to_bf16(const float* __restrict__ in,
                                                      unsigned short* __restrict__ out,
                                                      long long n) {
    long long i = (long long)blockIdx.x * blockDim.x + threadIdx.x;
    if (i < n) {
        float x = in[i];
        float g = 0.5f * x * (1.0f + erff(x * 0.70710678118654752440f));
        out[i] = f2bf(g);
    }
}

// ---------- transpose + f32->bf16: in[R][C] f32 -> out[C][R] bf16 ----------
// 32x32 tiles, 256 threads (32x8), coalesced both directions. R%32==0, C%32==0.
__global__ void __launch_bounds__(256)
k_transpose_bf16(const float* __restrict__ in, unsigned short* __restrict__ out,
                 int R, int C) {
    __shared__ float tile[32][33];
    const int tx = threadIdx.x & 31;
    const int ty = threadIdx.x >> 5;        // 0..7
    const int c0 = blockIdx.x * 32;
    const int r0 = blockIdx.y * 32;
#pragma unroll
    for (int j = 0; j < 32; j += 8)
        tile[ty + j][tx] = in[(size_t)(r0 + ty + j) * C + (c0 + tx)];
    __syncthreads();
#pragma unroll
    for (int j = 0; j < 32; j += 8)
        out[(size_t)(c0 + ty + j) * R + (r0 + tx)] = f2bf(tile[tx][ty + j]);
}

// ---------- LayerNorm (one block of 256 threads per row of 256) -> bf16 ----------
__global__ void __launch_bounds__(256) k_layernorm_bf16(const float* __restrict__ x,
                                                        const float* __restrict__ g,
                                                        const float* __restrict__ b,
                                                        unsigned short* __restrict__ out) {
    __shared__ float red[8];
    const int row = blockIdx.x;
    const int tid = threadIdx.x;
    const int wave = tid >> 5;
    const int lane = tid & 31;

    float v = x[(size_t)row * DMODEL + tid];

    float s = wave_sum32(v);
    if (lane == 0) red[wave] = s;
    __syncthreads();
    if (wave == 0) {
        float t = (lane < 8) ? red[lane] : 0.f;
        t = wave_sum32(t);
        if (lane == 0) red[0] = t;
    }
    __syncthreads();
    const float mu = red[0] * (1.0f / DMODEL);
    __syncthreads();

    float d = v - mu;
    float s2 = wave_sum32(d * d);
    if (lane == 0) red[wave] = s2;
    __syncthreads();
    if (wave == 0) {
        float t = (lane < 8) ? red[lane] : 0.f;
        t = wave_sum32(t);
        if (lane == 0) red[0] = t;
    }
    __syncthreads();
    const float var = red[0] * (1.0f / DMODEL);
    const float inv = rsqrtf(var + 1e-5f);
    const float y = d * inv * g[tid] + b[tid];
    out[(size_t)row * DMODEL + tid] = f2bf(y);
}

// ---------- bf16 WMMA GEMM: C[M,NC] = A[M,K](bf16,row) * Bt[NC,K](bf16) + bias ----------
// B is PRE-TRANSPOSED: Bt[n][k]. 256 threads = 8 waves (2x4); block tile 32x128;
// each wave: 16 rows x 32 cols (two 16x16 WMMA tiles sharing the A fragment).
// Ping-pong LDS double buffering -> one barrier per K step.
// Requires M%32==0, NC%128==0, K%32==0 (true for all calls here).
#define BLK_M 32
#define BLK_N 128
#define BLK_K 32
#define ASTR  40   // padded LDS row stride (halves); 80B rows, 16B aligned
#define BSTR  40

template <int NC, int K>
__global__ void __launch_bounds__(256)
k_gemm_bf16_wmma(const unsigned short* __restrict__ A,
                 const unsigned short* __restrict__ Bt,
                 const float* __restrict__ bias,
                 float* __restrict__ C) {
    __shared__ unsigned short As[2][BLK_M * ASTR];   // [m][k] padded
    __shared__ unsigned short Bs[2][BLK_N * BSTR];   // [n][k] padded

    const int tid  = threadIdx.x;
    const int wave = tid >> 5;
    const int lane = tid & 31;
    const int wr = wave >> 2;   // 0..1
    const int wc = wave & 3;    // 0..3
    const int blockM = blockIdx.y * BLK_M;
    const int blockN = blockIdx.x * BLK_N;

    // A staging: thread loads 4 contiguous halves (b64): row ar, cols ac4..+3
    const int ar  = tid >> 3;
    const int ac4 = (tid & 7) * 4;
    // B staging: thread loads 16 contiguous halves along K (2 x b128): row bn, cols bk16..+15
    const int bn   = tid >> 1;           // 0..127
    const int bk16 = (tid & 1) * 16;     // 0 or 16

    const unsigned short* aG = A  + (size_t)(blockM + ar) * K + ac4;
    const unsigned short* bG = Bt + (size_t)(blockN + bn) * K + bk16;

    v8f acc0 = {};
    v8f acc1 = {};

    int p = 0;
    for (int k0 = 0; k0 < K; k0 += BLK_K, p ^= 1) {
        // ---- stage A tile (b64 load + b64 LDS store) ----
        {
            ushort4 av = *(const ushort4*)(aG + k0);
            *(ushort4*)&As[p][ar * ASTR + ac4] = av;
        }
        // ---- stage B tile (2x b128 load + 2x b128 LDS store, no transpose needed) ----
        {
            uint4 b0 = *(const uint4*)(bG + k0);
            uint4 b1 = *(const uint4*)(bG + k0 + 8);
            *(uint4*)&Bs[p][bn * BSTR + bk16]     = b0;
            *(uint4*)&Bs[p][bn * BSTR + bk16 + 8] = b1;
        }
        if (k0 + BLK_K < K) {
            __builtin_prefetch(aG + (k0 + BLK_K), 0, 0);
            __builtin_prefetch(bG + (k0 + BLK_K), 0, 0);
        }
        __syncthreads();

        // ---- build fragments (all b128 LDS reads) ----
        union { uint4 q[2]; v16bf v; } afrag, bfrag0, bfrag1;
        {
            // A fragment, this code's contiguous k order: lanes 0-15 hold K{0..15}, lanes 16-31 K{16..31}
            const int m  = lane & 15;
            const int kg = lane >> 4;
            const unsigned short* arow = &As[p][(wr * 16 + m) * ASTR];
            afrag.q[0] = *(const uint4*)(arow + kg * 16);
            afrag.q[1] = *(const uint4*)(arow + kg * 16 + 8);
            // B fragment, same contiguous k order as A: lane = n + 16*(k/16); halves t -> k = 16*(lane/16)+t
            const int n  = lane & 15;
            const int kh = lane >> 4;
            const unsigned short* brow0 = &Bs[p][(wc * 32 + n) * BSTR + kh * 16];
            const unsigned short* brow1 = &Bs[p][(wc * 32 + 16 + n) * BSTR + kh * 16];
            bfrag0.q[0] = *(const uint4*)(brow0);
            bfrag0.q[1] = *(const uint4*)(brow0 + 8);
            bfrag1.q[0] = *(const uint4*)(brow1);
            bfrag1.q[1] = *(const uint4*)(brow1 + 8);
        }

        acc0 = __builtin_amdgcn_wmma_f32_16x16x32_bf16(false, afrag.v, false, bfrag0.v,
                                                       (short)0, acc0, false, false);
        acc1 = __builtin_amdgcn_wmma_f32_16x16x32_bf16(false, afrag.v, false, bfrag1.v,
                                                       (short)0, acc1, false, false);
        // no trailing barrier: next iteration writes the other LDS buffer; the
        // following iteration's barrier orders reuse of this one.
    }

    // C 16x16 f32 layout: VGPR r -> row r (lanes 0-15) / r+8 (lanes 16-31), lane%16 = col
    const int nlocal = lane & 15;
    const int mh = lane >> 4;
    const int col0 = blockN + wc * 32 + nlocal;
#pragma unroll
    for (int r = 0; r < 8; ++r) {
        const int row = blockM + wr * 16 + r + mh * 8;
        C[(size_t)row * NC + col0]      = acc0[r] + bias[col0];
        C[(size_t)row * NC + col0 + 16] = acc1[r] + bias[col0 + 16];
    }
}

// ---------- edge attention kernels ----------
__global__ void __launch_bounds__(256)
k_edge_logits(const float* __restrict__ qkv, const int* __restrict__ src,
              const int* __restrict__ dst, const float* __restrict__ dist,
              const float* __restrict__ path, float* __restrict__ logits) {
    long long i = (long long)blockIdx.x * blockDim.x + threadIdx.x;
    if (i >= (long long)E_EDGES * NHEADS) return;
    const int e = (int)(i >> 3);
    const int h = (int)(i & 7);
    const float4* qp = (const float4*)(qkv + (size_t)src[e] * (3 * DMODEL) + h * DHEAD);
    const float4* kp = (const float4*)(qkv + (size_t)dst[e] * (3 * DMODEL) + DMODEL + h * DHEAD);
    float s = 0.f;
#pragma unroll
    for (int t = 0; t < 8; ++t) {
        float4 a = qp[t], b = kp[t];
        s += a.x * b.x + a.y * b.y + a.z * b.z + a.w * b.w;
    }
    logits[i] = s * 0.0625f + dist[i] + path[i];  // scale = 256^-0.5
}

__global__ void __launch_bounds__(256)
k_seg_max(const float* __restrict__ logits, const int* __restrict__ dst,
          float* __restrict__ m) {
    long long i = (long long)blockIdx.x * blockDim.x + threadIdx.x;
    if (i >= (long long)E_EDGES * NHEADS) return;
    const int e = (int)(i >> 3);
    const int h = (int)(i & 7);
    atomicMaxF(&m[(size_t)dst[e] * NHEADS + h], logits[i]);
}

__global__ void __launch_bounds__(256)
k_exp_segsum(float* __restrict__ logits /* in: logits, out: p */,
             const int* __restrict__ dst, const float* __restrict__ m,
             float* __restrict__ z) {
    long long i = (long long)blockIdx.x * blockDim.x + threadIdx.x;
    if (i >= (long long)E_EDGES * NHEADS) return;
    const int e = (int)(i >> 3);
    const int h = (int)(i & 7);
    float p = expf(logits[i] - m[(size_t)dst[e] * NHEADS + h]);
    logits[i] = p;
    atomicAdd(&z[(size_t)dst[e] * NHEADS + h], p);
}

// one block per edge: src/dst are wave-uniform scalar loads
__global__ void __launch_bounds__(256)
k_aggregate(const float* __restrict__ qkv, const float* __restrict__ p,
            const float* __restrict__ z, const int* __restrict__ src,
            const int* __restrict__ dst, float* __restrict__ agg) {
    const int e = blockIdx.x;
    const int d = threadIdx.x;
    const int h = d >> 5;
    const int sd = dst[e];
    const int ss = src[e];
    float sa = p[(size_t)e * NHEADS + h] / z[(size_t)sd * NHEADS + h];
    float v = qkv[(size_t)ss * (3 * DMODEL) + 2 * DMODEL + d];
    atomicAdd(&agg[(size_t)sd * DMODEL + d], v * sa);
}

// ---------- launcher ----------
static inline int cdiv_ll(long long a, int b) { return (int)((a + b - 1) / b); }

extern "C" void kernel_launch(void* const* d_in, const int* in_sizes, int n_in,
                              void* d_out, int out_size, void* d_ws, size_t ws_size,
                              hipStream_t stream) {
    (void)in_sizes; (void)n_in; (void)out_size; (void)ws_size;

    const float* triplet  = (const float*)d_in[0];
    const float* dist     = (const float*)d_in[1];
    const float* path     = (const float*)d_in[2];
    const int*   src      = (const int*)d_in[3];
    const int*   dst      = (const int*)d_in[4];
    const float* ln1_g    = (const float*)d_in[5];
    const float* ln1_b    = (const float*)d_in[6];
    const float* qkv_w    = (const float*)d_in[7];
    const float* qkv_b    = (const float*)d_in[8];
    const float* rln_g    = (const float*)d_in[9];
    const float* rln_b    = (const float*)d_in[10];
    const float* inproj_w = (const float*)d_in[11];
    const float* inproj_b = (const float*)d_in[12];
    const float* ffn1_w   = (const float*)d_in[13];
    const float* ffn1_b   = (const float*)d_in[14];
    const float* ffn2_w   = (const float*)d_in[15];
    const float* ffn2_b   = (const float*)d_in[16];
    float* out = (float*)d_out;

    const long long ND  = (long long)N_NODES * DMODEL;       // 5.12M
    const long long N4D = (long long)N_NODES * 4 * DMODEL;   // 20.48M
    const long long NH  = (long long)N_NODES * NHEADS;
    const long long EH  = (long long)E_EDGES * NHEADS;

    size_t off = 0;
    auto suball = [&](size_t bytes) -> char* {
        char* p = (char*)d_ws + off;
        off += (bytes + 255) & ~(size_t)255;
        return p;
    };
    unsigned short* nh_bf   = (unsigned short*)suball(ND * 2);
    unsigned short* wqkv_t  = (unsigned short*)suball((size_t)DMODEL * 3 * DMODEL * 2);  // [768][256]
    unsigned short* winp_t  = (unsigned short*)suball((size_t)DMODEL * DMODEL * 2);      // [256][256]
    unsigned short* wf1_t   = (unsigned short*)suball((size_t)DMODEL * 4 * DMODEL * 2);  // [1024][256]
    unsigned short* wf2_t   = (unsigned short*)suball((size_t)4 * DMODEL * DMODEL * 2);  // [256][1024]
    float* qkv   = (float*)suball((size_t)N_NODES * 3 * DMODEL * 4);
    float* logit = (float*)suball((size_t)EH * 4);
    float* mbuf  = (float*)suball((size_t)NH * 4);
    float* zbuf  = (float*)suball((size_t)NH * 4);
    float* agg   = (float*)suball((size_t)ND * 4);
    unsigned short* agg_bf = (unsigned short*)suball(ND * 2);
    float* t1    = (float*)suball((size_t)ND * 4);   // inproj out
    float* xbuf  = (float*)suball((size_t)ND * 4);   // residual x
    unsigned short* yln_bf = (unsigned short*)suball(ND * 2);
    float* h1    = (float*)suball((size_t)N4D * 4);  // ffn1 out
    unsigned short* h1_bf  = (unsigned short*)suball(N4D * 2);

    // 1) weight transpose + f32->bf16 (w[K][N] -> wT[N][K])
    k_transpose_bf16<<<dim3(3 * DMODEL / 32, DMODEL / 32), 256, 0, stream>>>(qkv_w, wqkv_t, DMODEL, 3 * DMODEL);
    k_transpose_bf16<<<dim3(DMODEL / 32, DMODEL / 32), 256, 0, stream>>>(inproj_w, winp_t, DMODEL, DMODEL);
    k_transpose_bf16<<<dim3(4 * DMODEL / 32, DMODEL / 32), 256, 0, stream>>>(ffn1_w, wf1_t, DMODEL, 4 * DMODEL);
    k_transpose_bf16<<<dim3(DMODEL / 32, 4 * DMODEL / 32), 256, 0, stream>>>(ffn2_w, wf2_t, 4 * DMODEL, DMODEL);

    // 2) LN1 -> bf16
    k_layernorm_bf16<<<N_NODES, 256, 0, stream>>>(triplet, ln1_g, ln1_b, nh_bf);

    // 3) QKV GEMM: [20000,256] x [256,768] + bias -> qkv f32
    k_gemm_bf16_wmma<3 * DMODEL, DMODEL>
        <<<dim3(3 * DMODEL / BLK_N, N_NODES / BLK_M), 256, 0, stream>>>(nh_bf, wqkv_t, qkv_b, qkv);

    // 4) init segment buffers
    k_fill<<<cdiv_ll(NH, 256), 256, 0, stream>>>(mbuf, -INFINITY, NH);
    k_fill<<<cdiv_ll(NH, 256), 256, 0, stream>>>(zbuf, 0.0f, NH);
    k_fill<<<cdiv_ll(ND, 256), 256, 0, stream>>>(agg, 0.0f, ND);

    // 5) edge softmax + aggregation
    k_edge_logits<<<cdiv_ll(EH, 256), 256, 0, stream>>>(qkv, src, dst, dist, path, logit);
    k_seg_max<<<cdiv_ll(EH, 256), 256, 0, stream>>>(logit, dst, mbuf);
    k_exp_segsum<<<cdiv_ll(EH, 256), 256, 0, stream>>>(logit, dst, mbuf, zbuf);
    k_aggregate<<<E_EDGES, 256, 0, stream>>>(qkv, logit, zbuf, src, dst, agg);

    // 6) inproj GEMM + residual: x = triplet + (agg @ inproj_w + b)
    k_f32_to_bf16<<<cdiv_ll(ND, 256), 256, 0, stream>>>(agg, agg_bf, ND);
    k_gemm_bf16_wmma<DMODEL, DMODEL>
        <<<dim3(DMODEL / BLK_N, N_NODES / BLK_M), 256, 0, stream>>>(agg_bf, winp_t, inproj_b, t1);
    k_add<<<cdiv_ll(ND, 256), 256, 0, stream>>>(triplet, t1, xbuf, ND);

    // 7) FFN: y = gelu(LN(x) @ W1 + b1) @ W2 + b2 ; out = x + y
    k_layernorm_bf16<<<N_NODES, 256, 0, stream>>>(xbuf, rln_g, rln_b, yln_bf);
    k_gemm_bf16_wmma<4 * DMODEL, DMODEL>
        <<<dim3(4 * DMODEL / BLK_N, N_NODES / BLK_M), 256, 0, stream>>>(yln_bf, wf1_t, ffn1_b, h1);
    k_gelu_to_bf16<<<cdiv_ll(N4D, 256), 256, 0, stream>>>(h1, h1_bf, N4D);
    k_gemm_bf16_wmma<DMODEL, 4 * DMODEL>
        <<<dim3(DMODEL / BLK_N, N_NODES / BLK_M), 256, 0, stream>>>(h1_bf, wf2_t, ffn2_b, out);
    k_add_inplace<<<cdiv_ll(ND, 256), 256, 0, stream>>>(out, xbuf, ND);
}
